// PointNetSetAbstraction_59176059404777
// MI455X (gfx1250) — matrix operand order repaired and checked by reading, not compiled, not run on hardware
//
#include <hip/hip_runtime.h>
#include <hip/hip_bf16.h>
#include <math.h>

// ---------------------------------------------------------------------------
// Problem constants
// ---------------------------------------------------------------------------
#define BATCH   16
#define NPTS    4096
#define NPOINT  1024
#define NSAMPLE 32
#define CIN     64
#define M_ROWS  (BATCH * NPOINT * NSAMPLE)   // 524288
#define BN_EPS  1e-5f

typedef __attribute__((ext_vector_type(16))) _Float16 v16h;
typedef __attribute__((ext_vector_type(8)))  float    v8f;

union V16H { v16h h; uint4 u[2]; };
union H4   { _Float16 h[4]; uint2 u; };

// ---------------------------------------------------------------------------
// Workspace layout (bytes)
// ---------------------------------------------------------------------------
#define OFF_W16   ((size_t)0)                       // 16384 halves = 32768 B
#define OFF_STATS ((size_t)32768)                   // 768 f32 (sum/sumsq x3)
#define OFF_SS    ((size_t)36864)                   // 768 f32 (scale/shift x3)
#define OFF_X16A  ((size_t)65536)                   // M_ROWS*64 f16 = 67 MB
#define OFF_X16B  (OFF_X16A + (size_t)M_ROWS * 64 * 2)
#define OFF_Y     (OFF_X16B + (size_t)M_ROWS * 64 * 2)   // M_ROWS*128 f32 = 268 MB

// ---------------------------------------------------------------------------
// Kernel 0: convert weights f32->f16, zero BN accumulators (every call,
// since gemm kernels accumulate atomically).
// ---------------------------------------------------------------------------
__global__ __launch_bounds__(256) void prep_kernel(
    const float* __restrict__ W0, const float* __restrict__ W1,
    const float* __restrict__ W2, _Float16* __restrict__ W16,
    float* __restrict__ stats)
{
    int i = blockIdx.x * 256 + threadIdx.x;     // 0..16383
    float v;
    if (i < 4096)       v = W0[i];
    else if (i < 8192)  v = W1[i - 4096];
    else                v = W2[i - 8192];
    W16[i] = (_Float16)v;
    if (i < 768) stats[i] = 0.0f;
}

// ---------------------------------------------------------------------------
// Kernel 1: farthest point sampling. One block per batch; the 48 KB point
// cloud lives in LDS (WGP has 320 KB), so the 1024 dependent steps are pure
// LDS + block argmax. Emits new_xyz directly (reference semantics: emit the
// incoming 'far' each step, starting at index 0).
// ---------------------------------------------------------------------------
__global__ __launch_bounds__(256) void fps_kernel(
    const float* __restrict__ xyz, float* __restrict__ new_xyz)
{
    __shared__ float sx[NPTS], sy[NPTS], sz[NPTS];
    __shared__ float rv[256];
    __shared__ int   ri[256];
    __shared__ int   sfar;

    const int b = blockIdx.x;
    const int t = threadIdx.x;
    const float* base = xyz + (size_t)b * NPTS * 3;
    for (int i = t; i < NPTS; i += 256) {
        sx[i] = base[i * 3 + 0];
        sy[i] = base[i * 3 + 1];
        sz[i] = base[i * 3 + 2];
    }
    float dmin[16];
#pragma unroll
    for (int j = 0; j < 16; ++j) dmin[j] = 1e10f;
    if (t == 0) sfar = 0;
    __syncthreads();

    for (int s = 0; s < NPOINT; ++s) {
        const int far = sfar;
        if (t == 0) {
            float* o = new_xyz + ((size_t)b * NPOINT + s) * 3;
            o[0] = sx[far]; o[1] = sy[far]; o[2] = sz[far];
        }
        const float cx = sx[far], cy = sy[far], cz = sz[far];
        float bv = -1.0f; int bi = 0x7fffffff;
#pragma unroll
        for (int j = 0; j < 16; ++j) {
            const int i = t + j * 256;
            const float dx = sx[i] - cx, dy = sy[i] - cy, dz = sz[i] - cz;
            float d = dx * dx + dy * dy + dz * dz;
            d = fminf(d, dmin[j]);
            dmin[j] = d;
            if (d > bv || (d == bv && i < bi)) { bv = d; bi = i; }
        }
        rv[t] = bv; ri[t] = bi;
        __syncthreads();
        for (int off = 128; off > 0; off >>= 1) {
            if (t < off) {
                const float ov = rv[t + off]; const int oi = ri[t + off];
                if (ov > rv[t] || (ov == rv[t] && oi < ri[t])) { rv[t] = ov; ri[t] = oi; }
            }
            __syncthreads();
        }
        if (t == 0) sfar = ri[0];
        __syncthreads();
    }
}

// ---------------------------------------------------------------------------
// Kernel 2: 32-NN per center (argmin-with-invalidate over an LDS distance
// row) + gather neighbor features, converting to f16 for the WMMA GEMMs.
// One block per center; per-batch xyz (48 KB) is hammered by 1024 blocks and
// stays L2-resident.
// ---------------------------------------------------------------------------
__global__ __launch_bounds__(256) void knn_gather_kernel(
    const float* __restrict__ xyz, const float* __restrict__ points,
    const float* __restrict__ new_xyz, _Float16* __restrict__ Xh)
{
    __shared__ float dist[NPTS];
    __shared__ float rv[256];
    __shared__ int   ri[256];
    __shared__ int   knn[NSAMPLE];

    const int c = blockIdx.x;          // 0 .. BATCH*NPOINT-1
    const int b = c >> 10;
    const int t = threadIdx.x;

    const float* ctr = new_xyz + (size_t)c * 3;
    const float cx = ctr[0], cy = ctr[1], cz = ctr[2];
    const float* base = xyz + (size_t)b * NPTS * 3;

#pragma unroll
    for (int j = 0; j < 16; ++j) {
        const int i = t + j * 256;
        const float dx = base[i * 3 + 0] - cx;
        const float dy = base[i * 3 + 1] - cy;
        const float dz = base[i * 3 + 2] - cz;
        dist[i] = dx * dx + dy * dy + dz * dz;
    }
    __syncthreads();

    for (int k = 0; k < NSAMPLE; ++k) {
        float bv = 3.0e38f; int bi = 0x7fffffff;
#pragma unroll
        for (int j = 0; j < 16; ++j) {
            const int i = t + j * 256;
            const float d = dist[i];
            if (d < bv || (d == bv && i < bi)) { bv = d; bi = i; }
        }
        rv[t] = bv; ri[t] = bi;
        __syncthreads();
        for (int off = 128; off > 0; off >>= 1) {
            if (t < off) {
                const float ov = rv[t + off]; const int oi = ri[t + off];
                if (ov < rv[t] || (ov == rv[t] && oi < ri[t])) { rv[t] = ov; ri[t] = oi; }
            }
            __syncthreads();
        }
        if (t == 0) { knn[k] = ri[0]; dist[ri[0]] = 3.9e38f; }
        __syncthreads();
    }

    // Gather [NSAMPLE, CIN] features as f16
    const float* pb = points + (size_t)b * NPTS * CIN;
    _Float16* out = Xh + (size_t)c * NSAMPLE * CIN;
    for (int e = t; e < NSAMPLE * CIN; e += 256) {
        const int j = e >> 6, ch = e & 63;
        out[e] = (_Float16)pb[(size_t)knn[j] * CIN + ch];
    }
}

// ---------------------------------------------------------------------------
// Kernel 3: GEMM Y = X(f16)[M,64] * W(f16)^T + bias via v_wmma_f32_16x16x32_f16,
// with fused per-channel sum/sumsq accumulation for training-mode BN.
// 8 waves/block, each wave owns a 16-row strip; K=64 = two WMMA per N-tile.
// Register mapping per CDNA5 ISA §7.12.2.
// ---------------------------------------------------------------------------
__global__ __launch_bounds__(256) void gemm_bn_kernel(
    const _Float16* __restrict__ Xh, const _Float16* __restrict__ Wh,
    const float* __restrict__ bias, float* __restrict__ Y,
    float* __restrict__ gsum, float* __restrict__ gsq, int Cout)
{
    __shared__ float lsum[128], lsq[128];
    const int t = threadIdx.x;
    if (t < 128) { lsum[t] = 0.0f; lsq[t] = 0.0f; }
    __syncthreads();

    const int wave = t >> 5;
    const int lane = t & 31;
    const int row0 = blockIdx.x * 128 + wave * 16;
    const int ln = lane & 15;          // A: row m within tile; B: col n; C: col n
    const int lh = lane >> 4;          // lane-group select

    // A-matrix (16x32 f16): lane holds row ln; halves 0-7 = K in [8*lh, 8*lh+8),
    // halves 8-15 = same +16. Two 16-byte contiguous loads per K-chunk.
    const _Float16* arow = Xh + (size_t)(row0 + ln) * 64;
    V16H A0, A1;
    A0.u[0] = *(const uint4*)(arow + lh * 8);
    A0.u[1] = *(const uint4*)(arow + lh * 8 + 16);
    A1.u[0] = *(const uint4*)(arow + 32 + lh * 8);
    A1.u[1] = *(const uint4*)(arow + 32 + lh * 8 + 16);

    const int ntiles = Cout >> 4;
    for (int nt = 0; nt < ntiles; ++nt) {
        // B-matrix (32x16 f16): lane holds col n=ln; halves 0-7 = K in
        // [8*lh, 8*lh+8), halves 8-15 = same +16. B[k][n] = W[n][k] (row-major W[Cout][64]).
        const _Float16* wrow = Wh + (size_t)(nt * 16 + ln) * 64 + lh * 8;
        V16H B0, B1;
        B0.u[0] = *(const uint4*)(wrow);
        B0.u[1] = *(const uint4*)(wrow + 16);
        B1.u[0] = *(const uint4*)(wrow + 32);
        B1.u[1] = *(const uint4*)(wrow + 48);

        v8f acc = {};
        acc = __builtin_amdgcn_wmma_f32_16x16x32_f16(
            false, A0.h, false, B0.h, (short)0, acc, false, false);
        acc = __builtin_amdgcn_wmma_f32_16x16x32_f16(
            false, A1.h, false, B1.h, (short)0, acc, false, false);

        // C/D layout: VGPR r -> (m = r + 8*lh, n = ln)
        const int ch = nt * 16 + ln;
        const float bv = bias[ch];
        float s1 = 0.0f, s2 = 0.0f;
        float* yp = Y + (size_t)(row0 + 8 * lh) * (size_t)Cout + ch;
#pragma unroll
        for (int r = 0; r < 8; ++r) {
            const float v = acc[r] + bv;
            yp[(size_t)r * Cout] = v;
            s1 += v; s2 += v * v;
        }
        atomicAdd(&lsum[ch], s1);
        atomicAdd(&lsq[ch], s2);
    }
    __syncthreads();
    if (t < Cout) {
        atomicAdd(&gsum[t], lsum[t]);
        atomicAdd(&gsq[t], lsq[t]);
    }
}

// ---------------------------------------------------------------------------
// Kernel 4: fold channel sums into BN scale/shift.
// ---------------------------------------------------------------------------
__global__ void bn_scale_kernel(
    const float* __restrict__ gsum, const float* __restrict__ gsq,
    const float* __restrict__ g, const float* __restrict__ be,
    float* __restrict__ scale, float* __restrict__ shift, int Cout, float invM)
{
    const int t = threadIdx.x;
    if (t < Cout) {
        const float mu  = gsum[t] * invM;
        const float var = gsq[t] * invM - mu * mu;
        const float sc  = g[t] * rsqrtf(var + BN_EPS);
        scale[t] = sc;
        shift[t] = be[t] - mu * sc;
    }
}

// ---------------------------------------------------------------------------
// Kernel 5a/5b: normalize + ReLU. float4 in; f16 out (next layer input) or
// f32 out (final result). Cout is a power of two; grids divide exactly.
// ---------------------------------------------------------------------------
__global__ __launch_bounds__(256) void norm_relu_f16_kernel(
    const float* __restrict__ Y, const float* __restrict__ scale,
    const float* __restrict__ shift, _Float16* __restrict__ Xout, int Cout)
{
    const size_t idx = ((size_t)blockIdx.x * 256 + threadIdx.x) * 4;
    const float4 y = *(const float4*)(Y + idx);
    const int ch = (int)idx & (Cout - 1);
    H4 o;
    o.h[0] = (_Float16)fmaxf(0.0f, scale[ch + 0] * y.x + shift[ch + 0]);
    o.h[1] = (_Float16)fmaxf(0.0f, scale[ch + 1] * y.y + shift[ch + 1]);
    o.h[2] = (_Float16)fmaxf(0.0f, scale[ch + 2] * y.z + shift[ch + 2]);
    o.h[3] = (_Float16)fmaxf(0.0f, scale[ch + 3] * y.w + shift[ch + 3]);
    *(uint2*)(Xout + idx) = o.u;
}

__global__ __launch_bounds__(256) void norm_relu_f32_kernel(
    const float* __restrict__ Y, const float* __restrict__ scale,
    const float* __restrict__ shift, float* __restrict__ Out, int Cout)
{
    const size_t idx = ((size_t)blockIdx.x * 256 + threadIdx.x) * 4;
    const float4 y = *(const float4*)(Y + idx);
    const int ch = (int)idx & (Cout - 1);
    float4 o;
    o.x = fmaxf(0.0f, scale[ch + 0] * y.x + shift[ch + 0]);
    o.y = fmaxf(0.0f, scale[ch + 1] * y.y + shift[ch + 1]);
    o.z = fmaxf(0.0f, scale[ch + 2] * y.z + shift[ch + 2]);
    o.w = fmaxf(0.0f, scale[ch + 3] * y.w + shift[ch + 3]);
    *(float4*)(Out + idx) = o;
}

// ---------------------------------------------------------------------------
// Host-side launch sequence
// ---------------------------------------------------------------------------
extern "C" void kernel_launch(void* const* d_in, const int* in_sizes, int n_in,
                              void* d_out, int out_size, void* d_ws, size_t ws_size,
                              hipStream_t stream)
{
    const float* xyz    = (const float*)d_in[0];
    const float* points = (const float*)d_in[1];
    const float* W0 = (const float*)d_in[2];
    const float* b0 = (const float*)d_in[3];
    const float* g0 = (const float*)d_in[4];
    const float* be0 = (const float*)d_in[5];
    const float* W1 = (const float*)d_in[6];
    const float* b1 = (const float*)d_in[7];
    const float* g1 = (const float*)d_in[8];
    const float* be1 = (const float*)d_in[9];
    const float* W2 = (const float*)d_in[10];
    const float* b2 = (const float*)d_in[11];
    const float* g2 = (const float*)d_in[12];
    const float* be2 = (const float*)d_in[13];

    float* out = (float*)d_out;
    float* new_xyz = out;                                  // [16,1024,3]
    float* f_out   = out + (size_t)BATCH * NPOINT * 3;     // [16,1024,32,128]

    char* ws = (char*)d_ws;
    _Float16* W16   = (_Float16*)(ws + OFF_W16);
    float*    stats = (float*)(ws + OFF_STATS);            // layer l: sum @ l*256, sumsq @ l*256+128
    float*    ss    = (float*)(ws + OFF_SS);               // layer l: scale @ l*256, shift @ l*256+128
    _Float16* XA    = (_Float16*)(ws + OFF_X16A);
    _Float16* XB    = (_Float16*)(ws + OFF_X16B);
    float*    Yb    = (float*)(ws + OFF_Y);

    const float invM = 1.0f / (float)M_ROWS;

    prep_kernel<<<64, 256, 0, stream>>>(W0, W1, W2, W16, stats);
    fps_kernel<<<BATCH, 256, 0, stream>>>(xyz, new_xyz);
    knn_gather_kernel<<<BATCH * NPOINT, 256, 0, stream>>>(xyz, points, new_xyz, XA);

    // Layer 0: 64 -> 64
    gemm_bn_kernel<<<M_ROWS / 128, 256, 0, stream>>>(XA, W16, b0, Yb,
                                                     stats + 0, stats + 128, 64);
    bn_scale_kernel<<<1, 128, 0, stream>>>(stats + 0, stats + 128, g0, be0,
                                           ss + 0, ss + 128, 64, invM);
    norm_relu_f16_kernel<<<(M_ROWS * 64) / 1024, 256, 0, stream>>>(Yb, ss + 0, ss + 128, XB, 64);

    // Layer 1: 64 -> 64
    gemm_bn_kernel<<<M_ROWS / 128, 256, 0, stream>>>(XB, W16 + 4096, b1, Yb,
                                                     stats + 256, stats + 384, 64);
    bn_scale_kernel<<<1, 128, 0, stream>>>(stats + 256, stats + 384, g1, be1,
                                           ss + 256, ss + 384, 64, invM);
    norm_relu_f16_kernel<<<(M_ROWS * 64) / 1024, 256, 0, stream>>>(Yb, ss + 256, ss + 384, XA, 64);

    // Layer 2: 64 -> 128, f32 output to d_out
    gemm_bn_kernel<<<M_ROWS / 128, 256, 0, stream>>>(XA, W16 + 8192, b2, Yb,
                                                     stats + 512, stats + 640, 128);
    bn_scale_kernel<<<1, 128, 0, stream>>>(stats + 512, stats + 640, g2, be2,
                                           ss + 512, ss + 640, 128, invM);
    norm_relu_f32_kernel<<<(M_ROWS * 128) / 1024, 256, 0, stream>>>(Yb, ss + 512, ss + 640, f_out, 128);
}
